// MultiHeadAttention_76733885710480
// MI455X (gfx1250) — matrix operand order repaired and checked by reading, compiled, not run on hardware
//
#include <hip/hip_runtime.h>
#include <hip/hip_bf16.h>

#define DIM   2048
#define NH    32
#define NKV   8
#define HD    64
#define BATCH 2
#define SEQ   2048
#define BS    (BATCH * SEQ)   // 4096 rows
#define QCOLS (NH * HD)       // 2048
#define KVCOLS (NKV * HD)     // 512
#define SCALE 0.125f          // 64^-0.5
#define NEG_INF (-3.0e38f)

typedef __bf16 bf16_t;
typedef bf16_t v16bf __attribute__((ext_vector_type(16)));
typedef bf16_t v8bf  __attribute__((ext_vector_type(8)));
typedef float  v8f   __attribute__((ext_vector_type(8)));

__device__ __forceinline__ v8f wmma_bf16(v16bf a, v16bf b, v8f c) {
  return __builtin_amdgcn_wmma_f32_16x16x32_bf16(false, a, false, b, (short)0, c,
                                                 false, false);
}

// A-fragment: 16x32 bf16 tile, row-major, leading dim ld (2 x ds_load_b128).
__device__ __forceinline__ v16bf frag_a(const bf16_t* p, int ld, int lane) {
  int m  = lane & 15;
  int kh = (lane >> 4) << 4;  // 0 or 16
  const bf16_t* row = p + m * ld;
  v8bf lo = *(const v8bf*)(row + kh);
  v8bf hi = *(const v8bf*)(row + kh + 8);
  return __builtin_shufflevector(lo, hi, 0, 1, 2, 3, 4, 5, 6, 7,
                                 8, 9, 10, 11, 12, 13, 14, 15);
}

// B-fragment where memory holds elem(k,n) at p[n*ld + k] (N-major/transposed):
// lane's 16 K-values are contiguous -> one aligned 32B load (2 x b128).
__device__ __forceinline__ v16bf frag_bt(const bf16_t* p, int ld, int lane) {
  int n  = lane & 15;
  int kh = (lane >> 4) << 4;  // 0 or 16
  return *(const v16bf*)(p + n * ld + kh);
}

// B-fragment from swizzled GEMM weight tile Bsw[n][khalf][16].
__device__ __forceinline__ v16bf frag_b_swz(const bf16_t* base, int n0, int lane) {
  int n  = n0 + (lane & 15);
  int kh = lane >> 4;
  return *(const v16bf*)(base + (n * 2 + kh) * 16);
}

// ---------------------------------------------------------------------------
// Kernel 1 (x3 instantiations): projection x @ W (fp32 -> bf16 WMMA), optional
// RoPE epilogue. LDW is compile-time so weight-column loads use one base
// pointer + immediate offsets. Grid (BS/128, LDW/64); 128 threads = 4 waves;
// 128x64 block tile, 32 rows/wave, software-pipelined K loop.
// ---------------------------------------------------------------------------
template <int LDW, bool ROPE>
__global__ __launch_bounds__(128)
void proj_kernel(const float* __restrict__ x,
                 const float* __restrict__ W,
                 const float* __restrict__ cosT,
                 const float* __restrict__ sinT,
                 bf16_t* __restrict__ dst) {
  __shared__ bf16_t As[128][32];
  __shared__ bf16_t Bsw[64 * 2 * 16];  // [n 0..63][khalf 0..1][16 bf16]

  const int tid  = threadIdx.x;
  const int lane = tid & 31;
  const int w    = tid >> 5;
  const int mT   = blockIdx.x;
  const int wc0  = blockIdx.y * 64;

  v8f acc[2][4] = {{{}, {}, {}, {}}, {{}, {}, {}, {}}};

  const int bn = tid & 63, bkh = tid >> 6;

  float aF[32], bF[16];

  auto loadA = [&](int kk) {
    const float4* ap = (const float4*)(x + (size_t)(mT * 128 + tid) * DIM + kk);
#pragma unroll
    for (int i = 0; i < 8; ++i) {
      float4 f = ap[i];
      aF[4 * i + 0] = f.x; aF[4 * i + 1] = f.y;
      aF[4 * i + 2] = f.z; aF[4 * i + 3] = f.w;
    }
  };
  auto loadB = [&](int kk) {
    const float* base = W + (size_t)(kk + bkh * 16) * LDW + wc0 + bn;
#pragma unroll
    for (int i = 0; i < 16; ++i)
      bF[i] = base[(size_t)i * LDW];  // constant offsets -> immediate-offset loads
  };
  auto stage = [&]() {
#pragma unroll
    for (int c = 0; c < 4; ++c) {
      v8bf av;
#pragma unroll
      for (int i = 0; i < 8; ++i) av[i] = (bf16_t)aF[8 * c + i];
      *(v8bf*)&As[tid][8 * c] = av;
    }
    v8bf blo, bhi;
#pragma unroll
    for (int i = 0; i < 8; ++i) {
      blo[i] = (bf16_t)bF[i];
      bhi[i] = (bf16_t)bF[8 + i];
    }
    bf16_t* bd = &Bsw[(bn * 2 + bkh) * 16];
    *(v8bf*)bd       = blo;
    *(v8bf*)(bd + 8) = bhi;
  };

  loadA(0);
  loadB(0);
  for (int kk = 0; kk < DIM; kk += 32) {
    stage();
    __syncthreads();
    if (kk + 32 < DIM) { loadA(kk + 32); loadB(kk + 32); }  // overlap with WMMA
    v16bf a0 = frag_a(&As[32 * w][0], 32, lane);
    v16bf a1 = frag_a(&As[32 * w + 16][0], 32, lane);
#pragma unroll
    for (int t = 0; t < 4; ++t) {
      v16bf b = frag_b_swz(Bsw, 16 * t, lane);
      acc[0][t] = wmma_bf16(a0, b, acc[0][t]);
      acc[1][t] = wmma_bf16(a1, b, acc[1][t]);
    }
    __syncthreads();
  }

  // Epilogue: optional RoPE + bf16 store. C elem(m,n): m=j+8*(lane>=16), n=lane&15 (+16t)
  const int mrow = (lane >> 4) * 8;
  const int nloc = lane & 15;
#pragma unroll
  for (int r = 0; r < 2; ++r) {
#pragma unroll
    for (int j = 0; j < 8; ++j) {
      int rG  = mT * 128 + 32 * w + 16 * r + mrow + j;
      int pos = rG & (SEQ - 1);
      float vals[4];
      if (ROPE) {
#pragma unroll
        for (int t = 0; t < 4; ++t) {
          int d     = 16 * t + nloc;
          float c   = cosT[pos * HD + d];
          float s   = sinT[pos * HD + d];
          float cur = acc[r][t][j];
          float prt = (t < 2) ? -acc[r][t + 2][j] : acc[r][t - 2][j];
          vals[t]   = cur * c + prt * s;
        }
      } else {
#pragma unroll
        for (int t = 0; t < 4; ++t) vals[t] = acc[r][t][j];
      }
#pragma unroll
      for (int t = 0; t < 4; ++t)
        dst[(size_t)rG * LDW + wc0 + 16 * t + nloc] = (bf16_t)vals[t];
    }
  }
}

// ---------------------------------------------------------------------------
// Kernel 2: causal flash attention. Grid (SEQ/64, NH, BATCH), 128 threads.
// Q/K tiles staged with CDNA5 async-to-LDS DMA; V staged transposed.
// ---------------------------------------------------------------------------
__global__ __launch_bounds__(128)
void flash_attn_kernel(const bf16_t* __restrict__ qbuf,
                       const bf16_t* __restrict__ kbuf,
                       const bf16_t* __restrict__ vbuf,
                       bf16_t* __restrict__ attn) {
  __shared__ bf16_t Qs[64][64];
  __shared__ bf16_t Ks[64][64];
  __shared__ bf16_t Vt[64][64];     // transposed: [d][key]
  __shared__ bf16_t Ps[4][16][64];  // per-wave P scratch

  const int tid  = threadIdx.x;
  const int lane = tid & 31;
  const int w    = tid >> 5;
  const int qb   = blockIdx.x;
  const int h    = blockIdx.y;
  const int b    = blockIdx.z;
  const int kvh  = h >> 2;  // N_REP = 4

  const int crow = tid >> 1, cc0 = (tid & 1) * 32;  // copy thread mapping

  // --- stage Q tile via async DMA (global -> LDS, no VGPR round trip) ---
  {
    const bf16_t* src =
        qbuf + (size_t)(b * SEQ + qb * 64 + crow) * QCOLS + h * HD + cc0;
    unsigned ldsa = (unsigned)(uintptr_t)&Qs[crow][cc0];
#pragma unroll
    for (int i = 0; i < 4; ++i) {
      asm volatile("global_load_async_to_lds_b128 %0, %1, off"
                   :: "v"(ldsa + 16u * i), "v"((const void*)(src + 8 * i))
                   : "memory");
    }
  }
  asm volatile("s_wait_asynccnt 0" ::: "memory");
  __syncthreads();
  v16bf qfrag[2];
  qfrag[0] = frag_a(&Qs[16 * w][0], 64, lane);
  qfrag[1] = frag_a(&Qs[16 * w][32], 64, lane);

  v8f o[4] = {{}, {}, {}, {}};
  float rowMax[8], rowSum[8];
#pragma unroll
  for (int j = 0; j < 8; ++j) { rowMax[j] = NEG_INF; rowSum[j] = 0.0f; }

  const int mrow = (lane >> 4) * 8;
  const int nloc = lane & 15;
  const int vd = tid & 63, vkh = tid >> 6;  // V transpose staging mapping

  for (int kt = 0; kt <= qb; ++kt) {
    __syncthreads();  // previous K/V tiles fully consumed

    // K tile: async DMA copy (row-major [key][d])
    {
      const bf16_t* src =
          kbuf + (size_t)(b * SEQ + kt * 64 + crow) * KVCOLS + kvh * HD + cc0;
      unsigned ldsa = (unsigned)(uintptr_t)&Ks[crow][cc0];
#pragma unroll
      for (int i = 0; i < 4; ++i) {
        asm volatile("global_load_async_to_lds_b128 %0, %1, off"
                     :: "v"(ldsa + 16u * i), "v"((const void*)(src + 8 * i))
                     : "memory");
      }
    }
    // V tile: transpose while staging -> Vt[d][key]
    {
      size_t rb = (size_t)(b * SEQ + kt * 64 + vkh * 16) * KVCOLS + kvh * HD + vd;
      v8bf lo, hi;
#pragma unroll
      for (int i = 0; i < 8; ++i) {
        lo[i] = vbuf[rb + (size_t)i * KVCOLS];
        hi[i] = vbuf[rb + (size_t)(8 + i) * KVCOLS];
      }
      bf16_t* vdst = &Vt[vd][vkh * 16];
      *(v8bf*)vdst       = lo;
      *(v8bf*)(vdst + 8) = hi;
    }
    asm volatile("s_wait_asynccnt 0" ::: "memory");
    __syncthreads();

    // S = Q * K^T  (16 x 64 strip per wave)
    v8f s[4] = {{}, {}, {}, {}};
#pragma unroll
    for (int ks = 0; ks < 2; ++ks) {
#pragma unroll
      for (int t = 0; t < 4; ++t) {
        v16bf bk = frag_bt(&Ks[16 * t][ks * 32], 64, lane);
        s[t] = wmma_bf16(qfrag[ks], bk, s[t]);
      }
    }

    const bool diag = (kt == qb);
#pragma unroll
    for (int j = 0; j < 8; ++j) {
      int qrow = qb * 64 + 16 * w + mrow + j;
      float mx = NEG_INF;
#pragma unroll
      for (int t = 0; t < 4; ++t) {
        float v = s[t][j] * SCALE;
        if (diag) {
          int key = kt * 64 + 16 * t + nloc;
          if (key > qrow) v = NEG_INF;
        }
        s[t][j] = v;
        mx = fmaxf(mx, v);
      }
#pragma unroll
      for (int msk = 1; msk < 16; msk <<= 1)
        mx = fmaxf(mx, __shfl_xor(mx, msk, 32));
      float nm = fmaxf(rowMax[j], mx);
      float f  = __expf(rowMax[j] - nm);
      rowMax[j] = nm;
      float psum = 0.0f;
#pragma unroll
      for (int t = 0; t < 4; ++t) {
        float p = __expf(s[t][j] - nm);
        psum += p;
        Ps[w][mrow + j][16 * t + nloc] = (bf16_t)p;
      }
#pragma unroll
      for (int msk = 1; msk < 16; msk <<= 1)
        psum += __shfl_xor(psum, msk, 32);
      rowSum[j] = rowSum[j] * f + psum;
#pragma unroll
      for (int t = 0; t < 4; ++t) o[t][j] *= f;
    }

    // O += P * V   (V read through transposed layout -> contiguous frag loads)
#pragma unroll
    for (int ks = 0; ks < 2; ++ks) {
      v16bf ap = frag_a(&Ps[w][0][ks * 32], 64, lane);
#pragma unroll
      for (int dt = 0; dt < 4; ++dt) {
        v16bf bv = frag_bt(&Vt[16 * dt][ks * 32], 64, lane);
        o[dt] = wmma_bf16(ap, bv, o[dt]);
      }
    }
  }

#pragma unroll
  for (int j = 0; j < 8; ++j) {
    int qrow  = qb * 64 + 16 * w + mrow + j;
    float inv = 1.0f / rowSum[j];
#pragma unroll
    for (int t = 0; t < 4; ++t) {
      attn[(size_t)(b * SEQ + qrow) * QCOLS + h * HD + 16 * t + nloc] =
          (bf16_t)(o[t][j] * inv);
    }
  }
}

// ---------------------------------------------------------------------------
// Kernel 3: output projection attn(bf16) @ Wo(fp32->bf16) -> fp32 out.
// Grid (32, 32); 128x64 tile, 32 rows/wave, pipelined like kernel 1.
// ---------------------------------------------------------------------------
__global__ __launch_bounds__(128)
void out_proj_kernel(const bf16_t* __restrict__ attn,
                     const float* __restrict__ Wo,
                     float* __restrict__ out) {
  __shared__ bf16_t As[128][32];
  __shared__ bf16_t Bsw[64 * 2 * 16];

  const int tid  = threadIdx.x;
  const int lane = tid & 31;
  const int w    = tid >> 5;
  const int mT   = blockIdx.x;
  const int nG0  = blockIdx.y * 64;

  v8f acc[2][4] = {{{}, {}, {}, {}}, {{}, {}, {}, {}}};

  const int bn = tid & 63, bkh = tid >> 6;

  uint4 aQ[4];
  float bF[16];

  auto loadA = [&](int kk) {
    const uint4* ap = (const uint4*)(attn + (size_t)(mT * 128 + tid) * QCOLS + kk);
#pragma unroll
    for (int i = 0; i < 4; ++i) aQ[i] = ap[i];
  };
  auto loadB = [&](int kk) {
    const float* base = Wo + (size_t)(kk + bkh * 16) * DIM + nG0 + bn;
#pragma unroll
    for (int i = 0; i < 16; ++i)
      bF[i] = base[(size_t)i * DIM];  // constant offsets -> immediate-offset loads
  };
  auto stage = [&]() {
#pragma unroll
    for (int i = 0; i < 4; ++i) *(uint4*)&As[tid][8 * i] = aQ[i];
    v8bf blo, bhi;
#pragma unroll
    for (int i = 0; i < 8; ++i) {
      blo[i] = (bf16_t)bF[i];
      bhi[i] = (bf16_t)bF[8 + i];
    }
    bf16_t* bd = &Bsw[(bn * 2 + bkh) * 16];
    *(v8bf*)bd       = blo;
    *(v8bf*)(bd + 8) = bhi;
  };

  loadA(0);
  loadB(0);
  for (int kk = 0; kk < QCOLS; kk += 32) {
    stage();
    __syncthreads();
    if (kk + 32 < QCOLS) { loadA(kk + 32); loadB(kk + 32); }
    v16bf a0 = frag_a(&As[32 * w][0], 32, lane);
    v16bf a1 = frag_a(&As[32 * w + 16][0], 32, lane);
#pragma unroll
    for (int t = 0; t < 4; ++t) {
      v16bf b = frag_b_swz(Bsw, 16 * t, lane);
      acc[0][t] = wmma_bf16(a0, b, acc[0][t]);
      acc[1][t] = wmma_bf16(a1, b, acc[1][t]);
    }
    __syncthreads();
  }

  const int mrow = (lane >> 4) * 8;
  const int nloc = lane & 15;
#pragma unroll
  for (int r = 0; r < 2; ++r) {
#pragma unroll
    for (int j = 0; j < 8; ++j) {
      int rG = mT * 128 + 32 * w + 16 * r + mrow + j;
#pragma unroll
      for (int t = 0; t < 4; ++t)
        out[(size_t)rG * DIM + nG0 + 16 * t + nloc] = acc[r][t][j];
    }
  }
}

extern "C" void kernel_launch(void* const* d_in, const int* in_sizes, int n_in,
                              void* d_out, int out_size, void* d_ws, size_t ws_size,
                              hipStream_t stream) {
  const float* x    = (const float*)d_in[0];
  const float* Wq   = (const float*)d_in[1];
  const float* Wk   = (const float*)d_in[2];
  const float* Wv   = (const float*)d_in[3];
  const float* Wo   = (const float*)d_in[4];
  const float* cosT = (const float*)d_in[5];
  const float* sinT = (const float*)d_in[6];
  // d_in[7] = attention_mask: causal mask applied analytically in-kernel.
  float* out = (float*)d_out;

  char* ws = (char*)d_ws;
  bf16_t* qbuf = (bf16_t*)ws;                             // 4096*2048 bf16
  bf16_t* kbuf = (bf16_t*)(ws + (size_t)BS * QCOLS * 2);  // 4096*512 bf16
  bf16_t* vbuf = kbuf + (size_t)BS * KVCOLS;              // 4096*512 bf16
  bf16_t* attn = vbuf + (size_t)BS * KVCOLS;              // 4096*2048 bf16

  proj_kernel<QCOLS, true><<<dim3(BS / 128, QCOLS / 64), 128, 0, stream>>>(
      x, Wq, cosT, sinT, qbuf);
  proj_kernel<KVCOLS, true><<<dim3(BS / 128, KVCOLS / 64), 128, 0, stream>>>(
      x, Wk, cosT, sinT, kbuf);
  proj_kernel<KVCOLS, false><<<dim3(BS / 128, KVCOLS / 64), 128, 0, stream>>>(
      x, Wv, cosT, sinT, vbuf);
  flash_attn_kernel<<<dim3(SEQ / 64, NH, BATCH), 128, 0, stream>>>(
      qbuf, kbuf, vbuf, attn);
  out_proj_kernel<<<dim3(BS / 128, DIM / 64), 128, 0, stream>>>(attn, Wo, out);
}
